// ConvAct_37031208026361
// MI455X (gfx1250) — matrix operand order repaired and checked by reading, not compiled, not run on hardware
//
#include <hip/hip_runtime.h>

// ---------- types for WMMA (CDNA5 / gfx1250, wave32) ----------
typedef __attribute__((ext_vector_type(16))) __bf16 v16bf;
typedef __attribute__((ext_vector_type(8)))  __bf16 v8bf;
typedef __attribute__((ext_vector_type(8)))  float  v8f;

// ---------- problem constants ----------
#define CIN   128
#define COUT  256
#define HW    56
#define NIMG  32
#define SPAT  (HW * HW)          // 3136
#define NPOS  (NIMG * SPAT)      // 100352 = 784 * 128
#define BM    128                // Cout tile
#define BN    128                // spatial tile
#define TILES_N (NPOS / BN)      // 784
#define NITER 36                 // 9 taps * 4 chunks of K=32
#define EPS   1e-5f

// f32 -> bf16 (round-half-up) two at a time, packed with one v_perm_b32
__device__ __forceinline__ unsigned pack2bf(float a, float b) {
  unsigned ua = __builtin_bit_cast(unsigned, a) + 0x8000u;
  unsigned ub = __builtin_bit_cast(unsigned, b) + 0x8000u;
  // result bytes = { ub[3], ub[2], ua[3], ua[2] }  (high halves of both)
  return __builtin_amdgcn_perm(ub, ua, 0x07060302u);
}
__device__ __forceinline__ __bf16 f2bf(float f) {   // RNE, used once for weights
  unsigned u = __builtin_bit_cast(unsigned, f);
  u += 0x7FFFu + ((u >> 16) & 1u);
  unsigned short s = (unsigned short)(u >> 16);
  return __builtin_bit_cast(__bf16, s);
}

// ---------- kernel 0: zero BN accumulators (ws poisoned once; re-zero every launch) ----------
__global__ void k_zero(float* __restrict__ s) {
  s[threadIdx.x] = 0.0f;
  s[256 + threadIdx.x] = 0.0f;
}

// ---------- kernel 1: weights OIHW f32 -> bf16 [kk][Cout][Cin] ----------
__global__ void k_wcvt(const float* __restrict__ w, __bf16* __restrict__ wb) {
  const int idx = blockIdx.x * 256 + threadIdx.x;      // 0 .. 294911 exact
  const int kk   = idx / (COUT * CIN);
  const int rem  = idx % (COUT * CIN);
  const int cout = rem / CIN;
  const int c    = rem % CIN;
  wb[idx] = f2bf(w[(cout * CIN + c) * 9 + kk]);
}

// ---------- kernel 2: implicit-GEMM conv, bf16 WMMA, async weight staging, ping-pong LDS ----------
__global__ __launch_bounds__(256)
void k_conv(const float* __restrict__ x, const __bf16* __restrict__ wb,
            const float* __restrict__ bias, float* __restrict__ out,
            float* __restrict__ gsum, float* __restrict__ gsq) {
  __shared__ alignas(64) __bf16 Wt[2][BM * 32];   // A tiles (2 x 8 KB)
  __shared__ alignas(64) __bf16 At[2][BN * 32];   // B tiles (2 x 8 KB)
  __shared__ float sSum[BM], sSq[BM];

  const int tid  = threadIdx.x;
  const int lane = tid & 31;
  const int wv   = tid >> 5;        // 8 waves
  const int wm   = wv >> 1;         // 0..3 -> 32 rows of M
  const int wn   = wv & 1;          // 0..1 -> 64 cols of N

  const int m0 = (blockIdx.x / TILES_N) * BM;
  const int p0 = (blockIdx.x % TILES_N) * BN;

  if (tid < BM) { sSum[tid] = 0.0f; sSq[tid] = 0.0f; }

  v8f acc[2][4] = {};

  // staging roles
  const int wrow = tid >> 1, whalf = tid & 1;      // weight tile: row, 32B half
  const int spos = tid & 127;                      // act tile: spatial position
  const int cgrp = tid >> 7;                       // act tile: channel group (16 ch)
  const int p    = p0 + spos;
  const int nimg = p / SPAT;
  const int rr   = p % SPAT;
  const int hh   = rr / HW, ww = rr % HW;

  // ---- stagers (it = 0..35 -> kk = it/4, c0 = (it%4)*32) ----
  auto w_async = [&](int it, int buf) {
    const int kk = it >> 2, c0 = (it & 3) * 32;
    const __bf16* src = wb + ((kk * COUT + m0 + wrow) * CIN + c0 + whalf * 16);
    unsigned dst = (unsigned)(size_t)(void*)(&Wt[buf][wrow * 32 + whalf * 16]);
    // CDNA5 async global->LDS, tracked by ASYNCcnt; inst offset applies to
    // both LDS and memory addresses -> two issues move 32B/lane.
    asm volatile("global_load_async_to_lds_b128 %0, %1, off"
                 :: "v"(dst), "v"(src) : "memory");
    asm volatile("global_load_async_to_lds_b128 %0, %1, off offset:16"
                 :: "v"(dst), "v"(src) : "memory");
  };
  auto a_load = [&](int it, float* r) {
    const int kk = it >> 2, c0 = (it & 3) * 32;
    const int ky = kk / 3, kx = kk % 3;
    const int h2 = hh + ky - 1, w2 = ww + kx - 1;
    // clamp + mask instead of EXEC-predicated loads (keeps loads unconditional)
    const int h2c = min(max(h2, 0), HW - 1);
    const int w2c = min(max(w2, 0), HW - 1);
    const float mask =
        ((h2 >= 0) && (h2 < HW) && (w2 >= 0) && (w2 < HW)) ? 1.0f : 0.0f;
    const int xbase = ((nimg * CIN) * HW + h2c) * HW + w2c;
#pragma unroll
    for (int j = 0; j < 16; ++j)
      r[j] = x[xbase + (c0 + cgrp * 16 + j) * SPAT] * mask;
  };
  auto a_store = [&](const float* r, int buf) {
#pragma unroll
    for (int g = 0; g < 2; ++g) {
      uint4 u;
      u.x = pack2bf(r[g * 8 + 0], r[g * 8 + 1]);
      u.y = pack2bf(r[g * 8 + 2], r[g * 8 + 3]);
      u.z = pack2bf(r[g * 8 + 4], r[g * 8 + 5]);
      u.w = pack2bf(r[g * 8 + 6], r[g * 8 + 7]);
      *(uint4*)(&At[buf][spos * 32 + cgrp * 16 + g * 8]) = u;
    }
  };
  auto mma = [&](int buf) {
#pragma unroll
    for (int mi = 0; mi < 2; ++mi) {
      const int arow = wm * 32 + mi * 16 + (lane & 15);
      const int k0   = (lane < 16) ? 0 : 16;  // A: K 0-15 / 16-31
      v16bf a = *(const v16bf*)(&Wt[buf][arow * 32 + k0]);
#pragma unroll
      for (int ni = 0; ni < 4; ++ni) {
        const int brow = wn * 64 + ni * 16 + (lane & 15);
        const int bo   = (lane < 16) ? 0 : 16; // B: K 0-15 / 16-31
        v16bf b = *(const v16bf*)(&At[buf][brow * 32 + bo]);
        acc[mi][ni] = __builtin_amdgcn_wmma_f32_16x16x32_bf16(
            false, a, false, b, (short)0, acc[mi][ni], false, false);
      }
    }
  };

  // ---- software pipeline: stage chunk0, then overlap stage(i+1) with mma(i) ----
  {
    float r[16];
    a_load(0, r);
    w_async(0, 0);
    a_store(r, 0);
  }
  asm volatile("s_wait_asynccnt 0x0" ::: "memory");
  __syncthreads();

#pragma unroll 2
  for (int it = 0; it < NITER; ++it) {
    const int cur = it & 1;
    float r[16];
    if (it + 1 < NITER) {
      a_load(it + 1, r);        // global loads overlap the WMMAs below
      w_async(it + 1, cur ^ 1); // async engine fills the other weight buffer
    }
    mma(cur);
    if (it + 1 < NITER) a_store(r, cur ^ 1);
    asm volatile("s_wait_asynccnt 0x0" ::: "memory");
    __syncthreads();            // one barrier per K-chunk (ping-pong buffers)
  }

  // ---- epilogue: bias, store y, per-channel sum/sumsq ----
  const int col = lane & 15;
  const int mh  = (lane < 16) ? 0 : 8;   // C/D layout: lanes16-31 hold M+8
#pragma unroll
  for (int mi = 0; mi < 2; ++mi) {
#pragma unroll
    for (int ni = 0; ni < 4; ++ni) {
      const int pp  = p0 + wn * 64 + ni * 16 + col;
      const int ni2 = pp / SPAT, rr2 = pp % SPAT;
#pragma unroll
      for (int r = 0; r < 8; ++r) {
        const int cl = wm * 32 + mi * 16 + mh + r;         // 0..127 local channel
        const float y = acc[mi][ni][r] + bias[m0 + cl];
        out[(ni2 * COUT + (m0 + cl)) * SPAT + rr2] = y;
        atomicAdd(&sSum[cl], y);
        atomicAdd(&sSq[cl], y * y);
      }
    }
  }
  __syncthreads();
  if (tid < BM) {
    atomicAdd(&gsum[m0 + tid], sSum[tid]);
    atomicAdd(&gsq [m0 + tid], sSq [tid]);
  }
}

// ---------- kernel 3: fold stats into per-channel scale/shift ----------
__global__ void k_stats(const float* __restrict__ gsum, const float* __restrict__ gsq,
                        const float* __restrict__ gamma, const float* __restrict__ beta,
                        float* __restrict__ scale, float* __restrict__ shift) {
  const int c = threadIdx.x;
  const float inv  = 1.0f / (float)NPOS;
  const float mean = gsum[c] * inv;
  const float var  = gsq[c] * inv - mean * mean;
  const float sc   = gamma[c] * rsqrtf(var + EPS);
  scale[c] = sc;
  shift[c] = beta[c] - mean * sc;
}

// ---------- kernel 4: in-place BN + ReLU, float4 vectorized ----------
__global__ void k_bnrelu(float* __restrict__ out, const float* __restrict__ scale,
                         const float* __restrict__ shift) {
  const long i = ((long)blockIdx.x * 256 + threadIdx.x) * 4;
  const int  c = (int)((i / SPAT) % COUT);   // SPAT % 4 == 0 -> one channel per float4
  const float sc = scale[c], sh = shift[c];
  float4 v = *(float4*)(out + i);
  v.x = fmaxf(v.x * sc + sh, 0.0f);
  v.y = fmaxf(v.y * sc + sh, 0.0f);
  v.z = fmaxf(v.z * sc + sh, 0.0f);
  v.w = fmaxf(v.w * sc + sh, 0.0f);
  *(float4*)(out + i) = v;
}

extern "C" void kernel_launch(void* const* d_in, const int* in_sizes, int n_in,
                              void* d_out, int out_size, void* d_ws, size_t ws_size,
                              hipStream_t stream) {
  const float* x     = (const float*)d_in[0];
  const float* w     = (const float*)d_in[1];
  const float* bias  = (const float*)d_in[2];
  const float* gamma = (const float*)d_in[3];
  const float* beta  = (const float*)d_in[4];
  float* out = (float*)d_out;

  // workspace: sum[256] | sumsq[256] | scale[256] | shift[256] | bf16 weights
  float*  gsum  = (float*)d_ws;
  float*  gsq   = gsum + 256;
  float*  scale = gsum + 512;
  float*  shift = gsum + 768;
  __bf16* wb    = (__bf16*)((char*)d_ws + 4096);

  k_zero  <<<1,                      256, 0, stream>>>(gsum);
  k_wcvt  <<<(9 * COUT * CIN) / 256, 256, 0, stream>>>(w, wb);
  k_conv  <<<(COUT / BM) * TILES_N,  256, 0, stream>>>(x, wb, bias, out, gsum, gsq);
  k_stats <<<1,                      256, 0, stream>>>(gsum, gsq, gamma, beta, scale, shift);
  k_bnrelu<<<(NIMG * COUT * SPAT) / (256 * 4), 256, 0, stream>>>(out, scale, shift);
}
